// MolGraphNet_90580860273091
// MI455X (gfx1250) — compile-verified, matrix operand order checked by reading, not run on hardware
//
#include <hip/hip_runtime.h>
#include <hip/hip_bf16.h>

typedef __bf16 bf16;
typedef __attribute__((ext_vector_type(16))) __bf16 v16bf;
typedef __attribute__((ext_vector_type(8)))  __bf16 v8bf;
typedef __attribute__((ext_vector_type(8)))  float  v8f;

#define HID 256
#define HEADS 8
#define CH 32
#define EDIM 10
#define LAYERS 4
#define NEG_SLOPE 0.2f
static const float BN_S_H = 0.999995000037f; // 1/sqrt(1+1e-5)

// ---------- helpers ----------
__device__ __forceinline__ unsigned fenc(float f) {
    unsigned u = __float_as_uint(f);
    return (u & 0x80000000u) ? ~u : (u | 0x80000000u);
}
__device__ __forceinline__ float fdec(unsigned u) {
    unsigned b = (u & 0x80000000u) ? (u ^ 0x80000000u) : ~u;
    return __uint_as_float(b);
}

// ---------- generic WMMA GEMM: C[M][Nc] = act((A[M][K] @ Bt[Nc][K]^T + bias) * scale) ----------
// A, Bt in bf16; K in {96,128,256}; Nc multiple of 64. act: 0 none, 1 relu, 2 elu.
// B strip (64 cols x K) staged in LDS once per block; 8 waves share it. K-loop fully unrolled.
template<int K>
__global__ __launch_bounds__(256)
void k_wmma_gemm(const bf16* __restrict__ A, const bf16* __restrict__ Bt,
                 float* __restrict__ C, bf16* __restrict__ Cbf,
                 const float* __restrict__ bias,
                 int M, int Ncols, float scale, int act)
{
    constexpr int PITCH = K + 8;                 // +8 elem row pad (bank decorrelation)
    __shared__ bf16 Bs[64 * PITCH];
    const int wave = threadIdx.x >> 5;
    const int lane = threadIdx.x & 31;
    const int half = lane >> 4;                  // K-octet selector
    const int lr   = lane & 15;                  // row (A) / col (B,D) within tile
    const int rowBase = (blockIdx.x * 8 + wave) * 16;
    const int colBase = blockIdx.y * 64;

    // ---- cooperative stage of B strip [64 x K] into LDS (16B chunks) ----
    {
        constexpr int chunksPerRow = K >> 3;     // 8 bf16 per 16B chunk
        constexpr int totalChunks  = 64 * chunksPerRow;
#pragma unroll
        for (int base = 0; base < totalChunks; base += 256) {
            int idx = base + threadIdx.x;
            if (totalChunks % 256 == 0 || idx < totalChunks) {
                int r    = idx / chunksPerRow;
                int koff = (idx - r * chunksPerRow) << 3;
                v8bf v = *(const v8bf*)(Bt + (size_t)(colBase + r) * K + koff);
                *(v8bf*)(Bs + r * PITCH + koff) = v;
            }
        }
    }
    __syncthreads();

    if (rowBase < M) {                           // wave-uniform: EXEC all-ones inside
        v8f acc[4];
#pragma unroll
        for (int j = 0; j < 4; ++j)
#pragma unroll
            for (int i = 0; i < 8; ++i) acc[j][i] = 0.0f;

        int arow = rowBase + lr; if (arow >= M) arow = M - 1;   // clamp loads, guard stores
        const bf16* __restrict__ Arow = A + (size_t)arow * K;
        const bf16* __restrict__ BsL  = Bs + lr * PITCH + half * 8;

#pragma unroll
        for (int k0 = 0; k0 < K; k0 += 32) {
            // A fragment (16x32 bf16): VGPR0-3 = K[half*8..+7], VGPR4-7 = K[16+half*8..+7]
            v8bf alo = *(const v8bf*)(Arow + k0 + half * 8);
            v8bf ahi = *(const v8bf*)(Arow + k0 + 16 + half * 8);
            v16bf av;
#pragma unroll
            for (int i = 0; i < 8; ++i) { av[i] = alo[i]; av[i + 8] = ahi[i]; }
            // load all 4 B fragments of this k-step before issuing WMMAs
            v16bf bv[4];
#pragma unroll
            for (int j = 0; j < 4; ++j) {
                const bf16* brow = BsL + j * 16 * PITCH + k0;
                v8bf blo = *(const v8bf*)brow;          // ds_load_b128
                v8bf bhi = *(const v8bf*)(brow + 16);   // ds_load_b128
#pragma unroll
                for (int i = 0; i < 8; ++i) { bv[j][i] = blo[i]; bv[j][i + 8] = bhi[i]; }
            }
#pragma unroll
            for (int j = 0; j < 4; ++j)
                acc[j] = __builtin_amdgcn_wmma_f32_16x16x32_bf16(
                    false, av, false, bv[j], (short)0, acc[j], false, false);
        }

        // D layout: lane holds column (lr), VGPR v holds row v + 8*half
#pragma unroll
        for (int j = 0; j < 4; ++j) {
            const int col = colBase + j * 16 + lr;
            const float bc = bias ? bias[col] : 0.0f;
#pragma unroll
            for (int v = 0; v < 8; ++v) {
                const int row = rowBase + v + half * 8;
                if (row < M) {
                    float val = (acc[j][v] + bc) * scale;
                    if (act == 1)      val = val > 0.f ? val : 0.f;
                    else if (act == 2) val = val > 0.f ? val : (expf(val) - 1.f);
                    C[(size_t)row * Ncols + col] = val;
                    if (Cbf) Cbf[(size_t)row * Ncols + col] = (bf16)val;
                }
            }
        }
    }
}

// ---------- conversion / fill kernels ----------
__global__ void k_fill_u32(unsigned* __restrict__ p, unsigned v, size_t n) {
    size_t i = (size_t)blockIdx.x * blockDim.x + threadIdx.x;
    if (i < n) p[i] = v;
}
__global__ void k_f2bf(const float* __restrict__ in, bf16* __restrict__ out, size_t n) {
    size_t i = (size_t)blockIdx.x * blockDim.x + threadIdx.x;
    if (i < n) out[i] = (bf16)in[i];
}
// pad+convert row-major A[M][K] -> Abf[M][Kp]
__global__ void k_pad_convert(const float* __restrict__ A, bf16* __restrict__ Abf,
                              int M, int K, int Kp) {
    size_t i = (size_t)blockIdx.x * blockDim.x + threadIdx.x;
    size_t total = (size_t)M * Kp;
    if (i >= total) return;
    int n = (int)(i / Kp), k = (int)(i % Kp);
    Abf[i] = (k < K) ? (bf16)A[(size_t)n * K + k] : (bf16)0.0f;
}
// transpose+convert W[K][Nc] -> Wt[Nc][Kp] with zero pad in K
__global__ void k_transpose_bf16(const float* __restrict__ W, bf16* __restrict__ Wt,
                                 int K, int Nc, int Kp) {
    size_t i = (size_t)blockIdx.x * blockDim.x + threadIdx.x;
    size_t total = (size_t)Nc * Kp;
    if (i >= total) return;
    int n = (int)(i / Kp), k = (int)(i % Kp);
    Wt[i] = (k < K) ? (bf16)W[(size_t)k * Nc + n] : (bf16)0.0f;
}

// ---------- edge_attr column means (deterministic per-column block reduction) ----------
__global__ void k_ea_mean(const float* __restrict__ ea, float* __restrict__ mean, int E) {
    __shared__ float sm[256];
    const int d = blockIdx.x;   // 0..9
    float s = 0.f;
    for (int e = threadIdx.x; e < E; e += 256) s += ea[(size_t)e * EDIM + d];
    sm[threadIdx.x] = s; __syncthreads();
    for (int o = 128; o > 0; o >>= 1) {
        if (threadIdx.x < o) sm[threadIdx.x] += sm[threadIdx.x + o];
        __syncthreads();
    }
    if (threadIdx.x == 0) mean[d] = sm[0] / (float)E;
}

// ---------- edge attention: pass 1, logits + segment max ----------
__global__ void k_edge_logits(const float* __restrict__ xl, const float* __restrict__ xr,
                              const int* __restrict__ ei, const float* __restrict__ ea,
                              const float* __restrict__ eamean,
                              const float* __restrict__ We, const float* __restrict__ att,
                              float* __restrict__ logits, unsigned* __restrict__ nodemax,
                              int E, int Nn)
{
    int idx = blockIdx.x * blockDim.x + threadIdx.x;
    int total = (E + Nn) * HEADS;
    if (idx >= total) return;
    int e = idx >> 3, hh = idx & 7;
    int s, t; float eav[EDIM];
    if (e < E) {
        s = ei[e]; t = ei[E + e];
#pragma unroll
        for (int d = 0; d < EDIM; ++d) eav[d] = ea[(size_t)e * EDIM + d];
    } else {
        s = t = e - E;
#pragma unroll
        for (int d = 0; d < EDIM; ++d) eav[d] = eamean[d];
    }
    const float4* xls4 = (const float4*)(xl + (size_t)s * HID + hh * CH);
    const float4* xrt4 = (const float4*)(xr + (size_t)t * HID + hh * CH);
    const float* Wec = We + hh * CH;
    const float* ath = att + hh * CH;
    float lg = 0.f;
#pragma unroll
    for (int q = 0; q < CH / 4; ++q) {
        float4 a = xls4[q], b = xrt4[q];
        float av[4] = {a.x, a.y, a.z, a.w};
        float bv[4] = {b.x, b.y, b.z, b.w};
#pragma unroll
        for (int r = 0; r < 4; ++r) {
            int c = q * 4 + r;
            float ep = 0.f;
#pragma unroll
            for (int d = 0; d < EDIM; ++d) ep += eav[d] * Wec[d * HID + c];
            float v = av[r] + bv[r] + ep;
            v = v > 0.f ? v : NEG_SLOPE * v;
            lg += v * ath[c];
        }
    }
    logits[idx] = lg;
    atomicMax(&nodemax[(size_t)t * HEADS + hh], fenc(lg));
}

// ---------- edge attention: pass 2, exp + segment sum ----------
__global__ void k_edge_exp(const float* __restrict__ logits, const unsigned* __restrict__ nodemax,
                           const int* __restrict__ ei, float* __restrict__ exb,
                           float* __restrict__ denom, int E, int Nn)
{
    int idx = blockIdx.x * blockDim.x + threadIdx.x;
    int total = (E + Nn) * HEADS;
    if (idx >= total) return;
    int e = idx >> 3, hh = idx & 7;
    int t = (e < E) ? ei[E + e] : (e - E);
    float m = fdec(nodemax[(size_t)t * HEADS + hh]);
    float ex = expf(logits[idx] - m);
    exb[idx] = ex;
    atomicAdd(&denom[(size_t)t * HEADS + hh], ex);
}

// ---------- edge attention: pass 3, weighted aggregation ----------
__global__ void k_edge_agg(const float* __restrict__ exb, const float* __restrict__ denom,
                           const int* __restrict__ ei, const float* __restrict__ xl,
                           float* __restrict__ aggout, int E, int Nn)
{
    int idx = blockIdx.x * blockDim.x + threadIdx.x;
    int total = (E + Nn) * HEADS;
    if (idx >= total) return;
    int e = idx >> 3, hh = idx & 7;
    int s, t;
    if (e < E) { s = ei[e]; t = ei[E + e]; } else { s = t = e - E; }
    float alpha = exb[idx] / (denom[(size_t)t * HEADS + hh] + 1e-16f);
    const float* xls = xl + (size_t)s * HID + hh * CH;
    float* o = aggout + (size_t)t * HID + hh * CH;
#pragma unroll 8
    for (int c = 0; c < CH; ++c) atomicAdd(&o[c], alpha * xls[c]);
}

// ---------- per-layer update: h = elu((agg + b)*BN_S) + h ; also emit bf16 ----------
__global__ void k_layer_update(const float* __restrict__ agg, const float* __restrict__ bgat,
                               float* __restrict__ h, bf16* __restrict__ hbf, int Nn, float bns)
{
    size_t i = (size_t)blockIdx.x * blockDim.x + threadIdx.x;
    size_t total = (size_t)Nn * HID;
    if (i >= total) return;
    int c = (int)(i % HID);
    float v = (agg[i] + bgat[c]) * bns;
    v = v > 0.f ? v : (expf(v) - 1.f);
    v += h[i];
    h[i] = v;
    hbf[i] = (bf16)v;
}

// ---------- gate scalar + batch max ----------
__global__ void k_gate(const float* __restrict__ g, const float* __restrict__ Wg2,
                       const float* __restrict__ bg2, const int* __restrict__ batch,
                       float* __restrict__ gate, unsigned* __restrict__ bmax, int Nn)
{
    int n = blockIdx.x * blockDim.x + threadIdx.x;
    if (n >= Nn) return;
    float acc = bg2[0];
    const float* gr = g + (size_t)n * 128;
#pragma unroll 8
    for (int c = 0; c < 128; ++c) acc += gr[c] * Wg2[c];
    gate[n] = acc;
    atomicMax(&bmax[batch[n]], fenc(acc));
}
__global__ void k_gate_exp(const float* __restrict__ gate, const unsigned* __restrict__ bmax,
                           const int* __restrict__ batch, float* __restrict__ gex,
                           float* __restrict__ bden, int Nn)
{
    int n = blockIdx.x * blockDim.x + threadIdx.x;
    if (n >= Nn) return;
    int b = batch[n];
    float ex = expf(gate[n] - fdec(bmax[b]));
    gex[n] = ex;
    atomicAdd(&bden[b], ex);
}
__global__ void k_pool(const float* __restrict__ gex, const float* __restrict__ bden,
                       const int* __restrict__ batch, const float* __restrict__ h,
                       float* __restrict__ pooled, int Nn)
{
    int idx = blockIdx.x * blockDim.x + threadIdx.x;
    if (idx >= Nn * HEADS) return;
    int n = idx >> 3, hh = idx & 7;
    int b = batch[n];
    float coeff = gex[n] / (bden[b] + 1e-16f);
    const float* hr = h + (size_t)n * HID + hh * CH;
    float* o = pooled + (size_t)b * HID + hh * CH;
#pragma unroll 8
    for (int c = 0; c < CH; ++c) atomicAdd(&o[c], coeff * hr[c]);
}

// ---------- final: out = r3 @ W4 + b4 ----------
__global__ void k_final(const float* __restrict__ r3, const float* __restrict__ W4,
                        const float* __restrict__ b4, float* __restrict__ out, int Bg)
{
    int b = blockIdx.x * blockDim.x + threadIdx.x;
    if (b >= Bg) return;
    float acc = b4[0];
    const float* r = r3 + (size_t)b * 64;
#pragma unroll 8
    for (int c = 0; c < 64; ++c) acc += r[c] * W4[c];
    out[b] = acc;
}

// =====================================================================
extern "C" void kernel_launch(void* const* d_in, const int* in_sizes, int n_in,
                              void* d_out, int out_size, void* d_ws, size_t ws_size,
                              hipStream_t stream)
{
    const float* x         = (const float*)d_in[0];
    const int*   edge_index= (const int*)  d_in[1];
    const float* edge_attr = (const float*)d_in[2];
    const int*   batch     = (const int*)  d_in[3];
    const float* W_in = (const float*)d_in[4];  const float* b_in = (const float*)d_in[5];
    const float* Wl   = (const float*)d_in[6];  const float* Wr   = (const float*)d_in[7];
    const float* We   = (const float*)d_in[8];  const float* att  = (const float*)d_in[9];
    const float* b_gat= (const float*)d_in[10];
    const float* Wg1  = (const float*)d_in[11]; const float* bg1  = (const float*)d_in[12];
    const float* Wg2  = (const float*)d_in[13]; const float* bg2  = (const float*)d_in[14];
    const float* W1   = (const float*)d_in[15]; const float* b1   = (const float*)d_in[16];
    const float* W2   = (const float*)d_in[17]; const float* b2   = (const float*)d_in[18];
    const float* W3   = (const float*)d_in[19]; const float* b3   = (const float*)d_in[20];
    const float* W4   = (const float*)d_in[21]; const float* b4   = (const float*)d_in[22];

    const int Nn = in_sizes[3];              // 100000
    const int E  = in_sizes[2] / EDIM;       // 220000
    const int Bg = out_size;                 // 4096
    const int INdim = in_sizes[0] / Nn;      // 75
    const int Kp = ((INdim + 31) / 32) * 32; // 96
    const float bns = BN_S_H;

    // ---- workspace carve-out (256B aligned) ----
    char* wsb = (char*)d_ws; size_t off = 0;
    auto alloc = [&](size_t bytes) -> void* {
        void* p = wsb + off; off = (off + bytes + 255) & ~(size_t)255; return p;
    };
    bf16*  xpad   = (bf16*) alloc((size_t)Nn * Kp * 2);
    bf16*  WtIn   = (bf16*) alloc((size_t)HID * Kp * 2);
    bf16*  WtL    = (bf16*) alloc((size_t)LAYERS * HID * HID * 2);
    bf16*  WtR    = (bf16*) alloc((size_t)LAYERS * HID * HID * 2);
    bf16*  WtG1   = (bf16*) alloc((size_t)128 * HID * 2);
    bf16*  WtW1   = (bf16*) alloc((size_t)256 * 256 * 2);
    bf16*  WtW2   = (bf16*) alloc((size_t)128 * 256 * 2);
    bf16*  WtW3   = (bf16*) alloc((size_t)64 * 128 * 2);
    float* h      = (float*)alloc((size_t)Nn * HID * 4);
    bf16*  hbf    = (bf16*) alloc((size_t)Nn * HID * 2);
    float* xl     = (float*)alloc((size_t)Nn * HID * 4);
    float* xr     = (float*)alloc((size_t)Nn * HID * 4);
    float* logits = (float*)alloc((size_t)(E + Nn) * HEADS * 4);
    float* exb    = (float*)alloc((size_t)(E + Nn) * HEADS * 4);
    unsigned* nodemax = (unsigned*)alloc((size_t)Nn * HEADS * 4);
    float* denom  = (float*)alloc((size_t)Nn * HEADS * 4);
    float* aggout = (float*)alloc((size_t)Nn * HID * 4);
    float* eamean = (float*)alloc(EDIM * 4);
    float* g      = (float*)alloc((size_t)Nn * 128 * 4);
    float* gate   = (float*)alloc((size_t)Nn * 4);
    float* gex    = (float*)alloc((size_t)Nn * 4);
    unsigned* bmax= (unsigned*)alloc((size_t)Bg * 4);
    float* bden   = (float*)alloc((size_t)Bg * 4);
    float* pooled = (float*)alloc((size_t)Bg * HID * 4);
    bf16*  pooledbf=(bf16*) alloc((size_t)Bg * HID * 2);
    float* r1     = (float*)alloc((size_t)Bg * 256 * 4);
    bf16*  r1bf   = (bf16*) alloc((size_t)Bg * 256 * 2);
    float* r2     = (float*)alloc((size_t)Bg * 128 * 4);
    bf16*  r2bf   = (bf16*) alloc((size_t)Bg * 128 * 2);
    float* r3     = (float*)alloc((size_t)Bg * 64 * 4);
    (void)ws_size; (void)n_in;

    auto grid1 = [](size_t n) { return dim3((unsigned)((n + 255) / 256)); };
    auto gemm = [&](const bf16* A, const bf16* Bt, float* C, bf16* Cbf, const float* bias,
                    int M, int K, int Nc, float scale, int act) {
        dim3 grid((M + 127) / 128, Nc / 64);
        if (K == 96)
            k_wmma_gemm<96><<<grid, 256, 0, stream>>>(A, Bt, C, Cbf, bias, M, Nc, scale, act);
        else if (K == 128)
            k_wmma_gemm<128><<<grid, 256, 0, stream>>>(A, Bt, C, Cbf, bias, M, Nc, scale, act);
        else
            k_wmma_gemm<256><<<grid, 256, 0, stream>>>(A, Bt, C, Cbf, bias, M, Nc, scale, act);
    };

    // ---- weight prep (bf16, transposed, padded) ----
    k_pad_convert<<<grid1((size_t)Nn * Kp), 256, 0, stream>>>(x, xpad, Nn, INdim, Kp);
    k_transpose_bf16<<<grid1((size_t)HID * Kp), 256, 0, stream>>>(W_in, WtIn, INdim, HID, Kp);
    for (int i = 0; i < LAYERS; ++i) {
        k_transpose_bf16<<<grid1((size_t)HID * HID), 256, 0, stream>>>(
            Wl + (size_t)i * HID * HID, WtL + (size_t)i * HID * HID, HID, HID, HID);
        k_transpose_bf16<<<grid1((size_t)HID * HID), 256, 0, stream>>>(
            Wr + (size_t)i * HID * HID, WtR + (size_t)i * HID * HID, HID, HID, HID);
    }
    k_transpose_bf16<<<grid1((size_t)128 * HID), 256, 0, stream>>>(Wg1, WtG1, HID, 128, HID);
    k_transpose_bf16<<<grid1((size_t)256 * 256), 256, 0, stream>>>(W1, WtW1, 256, 256, 256);
    k_transpose_bf16<<<grid1((size_t)128 * 256), 256, 0, stream>>>(W2, WtW2, 256, 128, 256);
    k_transpose_bf16<<<grid1((size_t)64 * 128), 256, 0, stream>>>(W3, WtW3, 128, 64, 128);
    k_ea_mean<<<EDIM, 256, 0, stream>>>(edge_attr, eamean, E);

    // ---- input projection: h = elu((x@W_in + b_in) * BN_S) ----
    gemm(xpad, WtIn, h, hbf, b_in, Nn, Kp, HID, bns, /*elu*/2);

    // ---- GATv2 layers ----
    const size_t eTot = (size_t)(E + Nn) * HEADS;
    for (int i = 0; i < LAYERS; ++i) {
        gemm(hbf, WtL + (size_t)i * HID * HID, xl, nullptr, nullptr, Nn, HID, HID, 1.f, 0);
        gemm(hbf, WtR + (size_t)i * HID * HID, xr, nullptr, nullptr, Nn, HID, HID, 1.f, 0);
        k_fill_u32<<<grid1((size_t)Nn * HEADS), 256, 0, stream>>>(nodemax, 0u, (size_t)Nn * HEADS);
        k_fill_u32<<<grid1((size_t)Nn * HEADS), 256, 0, stream>>>((unsigned*)denom, 0u, (size_t)Nn * HEADS);
        k_fill_u32<<<grid1((size_t)Nn * HID), 256, 0, stream>>>((unsigned*)aggout, 0u, (size_t)Nn * HID);
        k_edge_logits<<<grid1(eTot), 256, 0, stream>>>(
            xl, xr, edge_index, edge_attr, eamean,
            We + (size_t)i * EDIM * HID, att + (size_t)i * HEADS * CH,
            logits, nodemax, E, Nn);
        k_edge_exp<<<grid1(eTot), 256, 0, stream>>>(logits, nodemax, edge_index, exb, denom, E, Nn);
        k_edge_agg<<<grid1(eTot), 256, 0, stream>>>(exb, denom, edge_index, xl, aggout, E, Nn);
        k_layer_update<<<grid1((size_t)Nn * HID), 256, 0, stream>>>(
            aggout, b_gat + (size_t)i * HID, h, hbf, Nn, bns);
    }

    // ---- GlobalAttention pooling ----
    gemm(hbf, WtG1, g, nullptr, bg1, Nn, HID, 128, 1.f, /*relu*/1);
    k_fill_u32<<<grid1((size_t)Bg), 256, 0, stream>>>(bmax, 0u, (size_t)Bg);
    k_fill_u32<<<grid1((size_t)Bg), 256, 0, stream>>>((unsigned*)bden, 0u, (size_t)Bg);
    k_fill_u32<<<grid1((size_t)Bg * HID), 256, 0, stream>>>((unsigned*)pooled, 0u, (size_t)Bg * HID);
    k_gate<<<grid1((size_t)Nn), 256, 0, stream>>>(g, Wg2, bg2, batch, gate, bmax, Nn);
    k_gate_exp<<<grid1((size_t)Nn), 256, 0, stream>>>(gate, bmax, batch, gex, bden, Nn);
    k_pool<<<grid1((size_t)Nn * HEADS), 256, 0, stream>>>(gex, bden, batch, h, pooled, Nn);
    k_f2bf<<<grid1((size_t)Bg * HID), 256, 0, stream>>>(pooled, pooledbf, (size_t)Bg * HID);

    // ---- readout MLP ----
    gemm(pooledbf, WtW1, r1, r1bf, b1, Bg, 256, 256, bns, 1);
    gemm(r1bf,     WtW2, r2, r2bf, b2, Bg, 256, 128, bns, 1);
    gemm(r2bf,     WtW3, r3, nullptr, b3, Bg, 128, 64, 1.f, 1);
    k_final<<<grid1((size_t)Bg), 256, 0, stream>>>(r3, W4, b4, (float*)d_out, Bg);
}
